// GTNLayer_90228672955082
// MI455X (gfx1250) — matrix operand order repaired and checked by reading, not compiled, not run on hardware
//
#include <hip/hip_runtime.h>
#include <stdint.h>

#define B_ROWS 50000
#define KN 32
#define DD 128
#define HH 4
#define DP 132   // padded neigh row stride in floats (TDM pads 4 dwords per 128)
#define HP 520   // padded hbuf row stride in bf16 elements

typedef __bf16 bf16_t;
typedef bf16_t v16bf __attribute__((ext_vector_type(16)));
typedef float  v8f   __attribute__((ext_vector_type(8)));
typedef unsigned v4u __attribute__((ext_vector_type(4)));
typedef unsigned v8u __attribute__((ext_vector_type(8)));

union V16U { uint4 q[2]; v16bf v; };

__device__ __forceinline__ unsigned short f2bf(float f) {
  unsigned u = __builtin_bit_cast(unsigned, f);
  unsigned r = 0x7FFFu + ((u >> 16) & 1u);
  return (unsigned short)((u + r) >> 16);
}

__device__ __forceinline__ unsigned pk2bf(float a, float b) {
  return (unsigned)f2bf(a) | ((unsigned)f2bf(b) << 16);
}

__device__ __forceinline__ float dot4(const float4& a, const float4& b) {
  return fmaf(a.x, b.x, fmaf(a.y, b.y, fmaf(a.z, b.z, a.w * b.w)));
}

__device__ __forceinline__ void fma4(float4& a, float s, const float4& v) {
  a.x = fmaf(s, v.x, a.x); a.y = fmaf(s, v.y, a.y);
  a.z = fmaf(s, v.z, a.z); a.w = fmaf(s, v.w, a.w);
}

__device__ __forceinline__ float bcast_lane(float v, int srcl) {
  return __builtin_bit_cast(float,
      __builtin_amdgcn_readlane(__builtin_bit_cast(int, v), srcl));
}

// ---------------------------------------------------------------------------
// TDM: issue one 2D tensor_load_to_lds for a 32x128 f32 tile.
//   tile_dim0 = 128 elems (512B rows), tile_dim1 = 32 rows,
//   tensor_dim0_stride = 128 elems, data_size = 4B,
//   pad_enable: +4 dwords after every 128 dwords -> LDS row stride = DP = 132.
// ---------------------------------------------------------------------------
__device__ __forceinline__ void tdm_issue_row(const float* gptr, unsigned ldsoff) {
  const unsigned long long ga = (unsigned long long)(const void*)gptr;
  const v4u g0 = {
      1u,                                        // count=1, is_restore=0, gather=0
      ldsoff,                                    // lds_addr  [63:32]
      (unsigned)ga,                              // global_addr[31:0]
      (unsigned)(ga >> 32) | (2u << 30)          // global_addr[56:32] | type=2
  };
  const v8u g1 = {
      (2u << 16) | (1u << 20) | (6u << 22) | (3u << 25),
      //  data_size=4B | pad_enable | pad_interval=128dw | pad_amount=4dw
      (unsigned)(128u << 16),                    // tensor_dim0[15:0] @ [63:48]
      (unsigned)(32u << 16),                     // tensor_dim1[15:0] @ [111:80]
      (unsigned)(128u << 16),                    // tile_dim0 @ [127:112]
      32u,                                       // tile_dim1 @ [143:128]
      128u,                                      // tensor_dim0_stride[31:0]
      0u, 0u
  };
  asm volatile("tensor_load_to_lds %0, %1" :: "s"(g0), "s"(g1) : "memory");
}

// ---------------------------------------------------------------------------
// Prep: W_proj f32 -> bf16 copy (B-fragment friendly: [n][k], k contiguous),
// and tbias[b][h] = tgt[b,:] . w_tgt[h,:] + b_heads[h]
// ---------------------------------------------------------------------------
__global__ __launch_bounds__(128) void gtn_prep(
    const float* __restrict__ tgt, const float* __restrict__ W_heads,
    const float* __restrict__ b_heads, const float* __restrict__ W_proj,
    unsigned short* __restrict__ wbf, float* __restrict__ tbias) {
  const int tid = blockIdx.x * blockDim.x + threadIdx.x;
  const int nth = gridDim.x * blockDim.x;
  for (int i = tid; i < DD * (HH * DD); i += nth) wbf[i] = f2bf(W_proj[i]);

  const int lane = tid & 31;
  const int wave = tid >> 5;
  const int nwave = nth >> 5;
  const float bh0 = b_heads[0], bh1 = b_heads[1], bh2 = b_heads[2], bh3 = b_heads[3];
  for (int b = wave; b < B_ROWS; b += nwave) {
    const float4 t4 = *(const float4*)(tgt + (size_t)b * DD + 4 * lane);
    float acc[HH];
#pragma unroll
    for (int h = 0; h < HH; ++h) {
      const float* w = W_heads + h * (2 * DD) + 4 * lane;  // w_tgt = W_heads[:, :D]
      acc[h] = fmaf(t4.x, w[0], fmaf(t4.y, w[1], fmaf(t4.z, w[2], t4.w * w[3])));
    }
#pragma unroll
    for (int h = 0; h < HH; ++h)
#pragma unroll
      for (int m = 16; m >= 1; m >>= 1) acc[h] += __shfl_xor(acc[h], m, 32);
    if (lane == 0) {
      tbias[(size_t)b * HH + 0] = acc[0] + bh0;
      tbias[(size_t)b * HH + 1] = acc[1] + bh1;
      tbias[(size_t)b * HH + 2] = acc[2] + bh2;
      tbias[(size_t)b * HH + 3] = acc[3] + bh3;
    }
  }
}

// ---------------------------------------------------------------------------
// Main fused kernel: one block (4 waves) per 16-row tile.
// Each wave double-buffers its rows through TDM.
// ---------------------------------------------------------------------------
#define BUF_F       (KN * DP)                     // 4224 floats per buffer
#define SMEM_BUF_F  (8 * BUF_F)                   // 4 waves x 2 buffers
#define SMEM_WNGH_F (HH * DD)                     // 512 floats
#define SMEM_BYTES  ((SMEM_BUF_F + SMEM_WNGH_F) * 4 + 16 * HP * 2)

__global__ __launch_bounds__(128) void gtn_main(
    const float* __restrict__ neigh, const float* __restrict__ W_heads,
    const float* __restrict__ tbias, const unsigned short* __restrict__ wbf,
    const float* __restrict__ b_proj, float* __restrict__ out) {
  __shared__ __align__(16) unsigned char smem[SMEM_BYTES];
  float* buf = (float*)smem;                       // LDS offset 0 (single shared var)
  float* wngh = buf + SMEM_BUF_F;
  unsigned short* hbuf = (unsigned short*)(wngh + SMEM_WNGH_F);

  const int t = threadIdx.x;
  const int lane = t & 31;
  const int wv = t >> 5;
  const int tile0 = blockIdx.x * 16;

  // preload w_ngh = W_heads[:, D:] as [h][d] f32
  for (int i = t; i < HH * DD; i += 128)
    wngh[i] = W_heads[(i >> 7) * (2 * DD) + DD + (i & (DD - 1))];
  __syncthreads();

  const unsigned lds0 = (unsigned)((wv * 2) * BUF_F * 4);  // this wave's buffer 0

  // prefetch first row tile for this wave
  tdm_issue_row(neigh + (size_t)(tile0 + wv) * (KN * DD), lds0);

#pragma unroll
  for (int rr = 0; rr < 4; ++rr) {
    const int cur = rr & 1;
    const int row = wv + rr * 4;
    const int b = tile0 + row;

    if (rr < 3) {  // prefetch next row tile into the other buffer
      tdm_issue_row(neigh + (size_t)(b + 4) * (KN * DD),
                    lds0 + (unsigned)(((rr + 1) & 1) * BUF_F * 4));
      asm volatile("s_wait_tensorcnt 0x1" ::: "memory");  // current tile landed
    } else {
      asm volatile("s_wait_tensorcnt 0x0" ::: "memory");
    }
    float* mybuf = buf + (wv * 2 + cur) * BUF_F;

    // ---- Stage A: score[k=lane][h] = neigh[k,:].w_ngh[h,:] + tbias[b][h] ----
    const float4 tb = *(const float4*)(tbias + (size_t)b * HH);
    float s0 = tb.x, s1 = tb.y, s2 = tb.z, s3 = tb.w;
    const float* nrow = mybuf + lane * DP;
#pragma unroll 8
    for (int c = 0; c < DD / 4; ++c) {
      const float4 nv = *(const float4*)(nrow + 4 * c);
      const float4 w0 = *(const float4*)(wngh + 0 * DD + 4 * c);
      const float4 w1 = *(const float4*)(wngh + 1 * DD + 4 * c);
      const float4 w2 = *(const float4*)(wngh + 2 * DD + 4 * c);
      const float4 w3 = *(const float4*)(wngh + 3 * DD + 4 * c);
      s0 += dot4(nv, w0); s1 += dot4(nv, w1);
      s2 += dot4(nv, w2); s3 += dot4(nv, w3);
    }
    // relu + softmax over k (= lanes of this wave32)
    float z0 = fmaxf(s0, 0.f), z1 = fmaxf(s1, 0.f);
    float z2 = fmaxf(s2, 0.f), z3 = fmaxf(s3, 0.f);
    float m0 = z0, m1 = z1, m2 = z2, m3 = z3;
#pragma unroll
    for (int m = 16; m >= 1; m >>= 1) {
      m0 = fmaxf(m0, __shfl_xor(m0, m, 32));
      m1 = fmaxf(m1, __shfl_xor(m1, m, 32));
      m2 = fmaxf(m2, __shfl_xor(m2, m, 32));
      m3 = fmaxf(m3, __shfl_xor(m3, m, 32));
    }
    float e0 = __expf(z0 - m0), e1 = __expf(z1 - m1);
    float e2 = __expf(z2 - m2), e3 = __expf(z3 - m3);
    float q0 = e0, q1 = e1, q2 = e2, q3 = e3;
#pragma unroll
    for (int m = 16; m >= 1; m >>= 1) {
      q0 += __shfl_xor(q0, m, 32); q1 += __shfl_xor(q1, m, 32);
      q2 += __shfl_xor(q2, m, 32); q3 += __shfl_xor(q3, m, 32);
    }
    const float a0 = e0 / q0, a1 = e1 / q1, a2 = e2 / q2, a3 = e3 / q3;

    // ---- Stage B: h[h][d] = sum_k att[k][h] * neigh[k][d]; lane owns d=4L..4L+3
    float4 acc0 = {0, 0, 0, 0}, acc1 = {0, 0, 0, 0};
    float4 acc2 = {0, 0, 0, 0}, acc3 = {0, 0, 0, 0};
#pragma unroll
    for (int k = 0; k < KN; ++k) {
      const float4 nv = *(const float4*)(mybuf + k * DP + 4 * lane);
      const float b0 = bcast_lane(a0, k);
      const float b1 = bcast_lane(a1, k);
      const float b2 = bcast_lane(a2, k);
      const float b3 = bcast_lane(a3, k);
      fma4(acc0, b0, nv); fma4(acc1, b1, nv);
      fma4(acc2, b2, nv); fma4(acc3, b3, nv);
    }
    // store h tile row as bf16: hbuf[row][h*128 + d]
    unsigned short* hrow = hbuf + row * HP;
    *(unsigned*)(hrow + 0 * DD + 4 * lane + 0) = pk2bf(acc0.x, acc0.y);
    *(unsigned*)(hrow + 0 * DD + 4 * lane + 2) = pk2bf(acc0.z, acc0.w);
    *(unsigned*)(hrow + 1 * DD + 4 * lane + 0) = pk2bf(acc1.x, acc1.y);
    *(unsigned*)(hrow + 1 * DD + 4 * lane + 2) = pk2bf(acc1.z, acc1.w);
    *(unsigned*)(hrow + 2 * DD + 4 * lane + 0) = pk2bf(acc2.x, acc2.y);
    *(unsigned*)(hrow + 2 * DD + 4 * lane + 2) = pk2bf(acc2.z, acc2.w);
    *(unsigned*)(hrow + 3 * DD + 4 * lane + 0) = pk2bf(acc3.x, acc3.y);
    *(unsigned*)(hrow + 3 * DD + 4 * lane + 2) = pk2bf(acc3.z, acc3.w);
  }
  __syncthreads();

  // ---- Stage C: out[16 x 128] = hbuf[16 x 512] @ W_proj^T + b_proj (WMMA bf16)
  {
    const int half = (lane < 16) ? 0 : 8;
    const int ln = lane & 15;
    const int n0 = wv * 32;                  // this wave's two 16-col tiles
    const float bp0 = b_proj[n0 + ln];
    const float bp1 = b_proj[n0 + 16 + ln];
    v8f c0 = {bp0, bp0, bp0, bp0, bp0, bp0, bp0, bp0};
    v8f c1 = {bp1, bp1, bp1, bp1, bp1, bp1, bp1, bp1};
    const unsigned short* arow_p = hbuf + ln * HP;   // A: M = lane%16
    const int aoff = (lane < 16) ? 0 : 16;           // 16-bit A frag chunking
    const int boff = (lane < 16) ? 0 : 16;           // 16-bit B frag chunking
#pragma unroll
    for (int j = 0; j < 16; ++j) {                   // K = 512 in steps of 32
      V16U a, b0u, b1u;
      const int ka = 32 * j + aoff;
      a.q[0] = *(const uint4*)(arow_p + ka);
      a.q[1] = *(const uint4*)(arow_p + ka + 8);
      const int kb = 32 * j + boff;
      const unsigned short* w0p = wbf + (size_t)(n0 + ln) * 512 + kb;
      const unsigned short* w1p = wbf + (size_t)(n0 + 16 + ln) * 512 + kb;
      b0u.q[0] = *(const uint4*)(w0p);
      b0u.q[1] = *(const uint4*)(w0p + 8);
      b1u.q[0] = *(const uint4*)(w1p);
      b1u.q[1] = *(const uint4*)(w1p + 8);
      c0 = __builtin_amdgcn_wmma_f32_16x16x32_bf16(false, a.v, false, b0u.v,
                                                   (short)0, c0, false, false);
      c1 = __builtin_amdgcn_wmma_f32_16x16x32_bf16(false, a.v, false, b1u.v,
                                                   (short)0, c1, false, false);
    }
#pragma unroll
    for (int j = 0; j < 8; ++j) {
      const int M = j + half;                        // C/D frag row mapping
      float* orow = out + (size_t)(tile0 + M) * DD;
      orow[n0 + ln] = c0[j];
      orow[n0 + 16 + ln] = c1[j];
    }
  }
}

// ---------------------------------------------------------------------------
extern "C" void kernel_launch(void* const* d_in, const int* in_sizes, int n_in,
                              void* d_out, int out_size, void* d_ws, size_t ws_size,
                              hipStream_t stream) {
  const float* tgt     = (const float*)d_in[0];
  const float* neigh   = (const float*)d_in[1];
  const float* W_heads = (const float*)d_in[2];
  const float* b_heads = (const float*)d_in[3];
  const float* W_proj  = (const float*)d_in[4];
  const float* b_proj  = (const float*)d_in[5];
  float* out = (float*)d_out;

  unsigned short* wbf = (unsigned short*)d_ws;                  // 128*512*2 B
  float* tbias = (float*)((char*)d_ws + (size_t)(DD * HH * DD) * 2);  // B*4 f32

  gtn_prep<<<400, 128, 0, stream>>>(tgt, W_heads, b_heads, W_proj, wbf, tbias);
  gtn_main<<<B_ROWS / 16, 128, 0, stream>>>(neigh, W_heads, tbias, wbf, b_proj, out);
}
